// Attention_1357209666513
// MI455X (gfx1250) — matrix operand order repaired and checked by reading, not compiled, not run on hardware
//
#include <hip/hip_runtime.h>
#include <hip/hip_bf16.h>

#define B_    2
#define S_    2048
#define HID_  1024
#define NH_   16
#define NKV_  4
#define HD_   64

typedef __attribute__((ext_vector_type(16))) _Float16     v16h;
typedef __attribute__((ext_vector_type(8)))  _Float16     hv8;
typedef __attribute__((ext_vector_type(8)))  float        v8f;
typedef __attribute__((ext_vector_type(4)))  unsigned int u32x4;
typedef __attribute__((ext_vector_type(8)))  int          i32x8;
typedef __attribute__((ext_vector_type(4)))  int          i32x4;

#if __has_builtin(__builtin_amdgcn_tensor_load_to_lds) && \
    __has_builtin(__builtin_amdgcn_s_wait_tensorcnt)
#define USE_TDM 1
#else
#define USE_TDM 0
#endif

#define WMMA_F16(Aop, Bop, Cop) \
  __builtin_amdgcn_wmma_f32_16x16x32_f16(false, (Aop), false, (Bop), (short)0, (Cop), false, false)

// Concatenate the two 8-half runs of an A-operand (k = 16g + 0..15).
#define CAT16(lo, hi) __builtin_shufflevector((lo), (hi), 0,1,2,3,4,5,6,7,8,9,10,11,12,13,14,15)

// -------------------------------------------------------------------------
// Cross-lane reductions within each 16-lane group (wave32; xor masks 1,2,4,8
// stay inside a group since the group bit is bit4).
// -------------------------------------------------------------------------
__device__ __forceinline__ float group_max(float t) {
  t = fmaxf(t, __shfl_xor(t, 1, 32));
  t = fmaxf(t, __shfl_xor(t, 2, 32));
  t = fmaxf(t, __shfl_xor(t, 4, 32));
  t = fmaxf(t, __shfl_xor(t, 8, 32));
  return t;
}
__device__ __forceinline__ float group_sum(float t) {
  t += __shfl_xor(t, 1, 32);
  t += __shfl_xor(t, 2, 32);
  t += __shfl_xor(t, 4, 32);
  t += __shfl_xor(t, 8, 32);
  return t;
}

__device__ __forceinline__ hv8 cvt8(const float4 a, const float4 b) {
  hv8 h;
  h[0] = (_Float16)a.x; h[1] = (_Float16)a.y; h[2] = (_Float16)a.z; h[3] = (_Float16)a.w;
  h[4] = (_Float16)b.x; h[5] = (_Float16)b.y; h[6] = (_Float16)b.z; h[7] = (_Float16)b.w;
  return h;
}

// -------------------------------------------------------------------------
// GEMM: C[M,N] = A[M,K] @ W[N,K]^T  (fp32 in memory, f16 WMMA compute).
// Block = 128 threads = 4 waves computing a 64(M) x 64(N) tile; wave w owns
// rows [w*16, w*16+16) x all 64 N (4 WMMA per K-step, A-operand reused 4x).
// Epilogue stores f32 (final proj) or f16 (Q/K/V workspace).
// -------------------------------------------------------------------------
__global__ __launch_bounds__(128) void gemm_wmma_f16(
    const float* __restrict__ A, const float* __restrict__ W,
    void* __restrict__ Cout, int M, int N, int K, int store_half)
{
  __shared__ _Float16 a_lds[64][32];   // [m][k]
  __shared__ _Float16 b_lds[64][32];   // [n][k]  == W row-major tile

  const int tid  = threadIdx.x;
  const int wave = tid >> 5;
  const int lane = tid & 31;
  const int g    = lane >> 4;
  const int l15  = lane & 15;

  const int m0 = blockIdx.y * 64;
  const int n0 = blockIdx.x * 64;

  v8f acc[4] = {};

  for (int k0 = 0; k0 < K; k0 += 32) {
    // --- stage A and W tiles (64x32 each), fp32 -> f16, 8 elems/slot ---
#pragma unroll
    for (int s = 0; s < 2; ++s) {
      const int slot = tid + s * 128;       // 0..255
      const int r = slot >> 2;              // row 0..63
      const int q = (slot & 3) * 8;         // k-offset 0/8/16/24
      const float* ap = A + (size_t)(m0 + r) * K + k0 + q;
      const float* wp = W + (size_t)(n0 + r) * K + k0 + q;
      const float4 a0 = *(const float4*)(ap);
      const float4 a1 = *(const float4*)(ap + 4);
      const float4 w0 = *(const float4*)(wp);
      const float4 w1 = *(const float4*)(wp + 4);
      *(hv8*)&a_lds[r][q] = cvt8(a0, a1);
      *(hv8*)&b_lds[r][q] = cvt8(w0, w1);
      if (k0 + 32 < K) {
        __builtin_prefetch(ap + 32, 0, 3);
        __builtin_prefetch(wp + 32, 0, 3);
      }
    }
    __syncthreads();

    // --- A operand for this wave: row = wave*16 + l15, two 16B runs ---
    const int ar = wave * 16 + l15;
    const hv8 alo = *(const hv8*)&a_lds[ar][16 * g];
    const hv8 ahi = *(const hv8*)&a_lds[ar][16 * g + 8];
    const v16h a  = CAT16(alo, ahi);

    // --- 4 WMMAs across the 64-wide N tile, B operand = contiguous 32B ---
#pragma unroll
    for (int nt = 0; nt < 4; ++nt) {
      const v16h b = *(const v16h*)&b_lds[nt * 16 + l15][16 * g];
      acc[nt] = WMMA_F16(a, b, acc[nt]);
    }
    __syncthreads();
  }

  if (store_half) {
    _Float16* C = (_Float16*)Cout;
#pragma unroll
    for (int nt = 0; nt < 4; ++nt)
#pragma unroll
      for (int v = 0; v < 8; ++v)
        C[(size_t)(m0 + wave * 16 + v + 8 * g) * N + n0 + nt * 16 + l15] =
            (_Float16)acc[nt][v];
  } else {
    float* C = (float*)Cout;
#pragma unroll
    for (int nt = 0; nt < 4; ++nt)
#pragma unroll
      for (int v = 0; v < 8; ++v)
        C[(size_t)(m0 + wave * 16 + v + 8 * g) * N + n0 + nt * 16 + l15] =
            acc[nt][v];
  }
}

// -------------------------------------------------------------------------
// RoPE in-place on f16 data: each thread owns pair (d, d+32) of one head.
// -------------------------------------------------------------------------
__global__ void rope_kernel(_Float16* __restrict__ t,
                            const float* __restrict__ cosT,
                            const float* __restrict__ sinT,
                            int H, int total)
{
  const int idx = blockIdx.x * blockDim.x + threadIdx.x;
  if (idx >= total) return;
  const int d  = idx & 31;
  const int h  = (idx >> 5) % H;
  const int bs = idx / (32 * H);
  const int s  = bs % S_;
  const size_t base = ((size_t)bs * H + h) * (size_t)HD_;
  const float x1 = (float)t[base + d];
  const float x2 = (float)t[base + d + 32];
  const float cv = cosT[s * HD_ + d];
  const float sv = sinT[s * HD_ + d];
  t[base + d]      = (_Float16)(x1 * cv - x2 * sv);
  t[base + d + 32] = (_Float16)(x2 * cv + x1 * sv);
}

// -------------------------------------------------------------------------
// Flash attention: one wave per (batch, q-head, 16-row Q tile). All Q/K/V
// already f16. K tile (32 keys x 64 d, row-major, X-fastest) is moved by the
// Tensor Data Mover straight into LDS (2-D D# descriptor, data_size=2B,
// tile 64x32, row stride NKV*HD), synced with s_wait_tensorcnt. V staged
// manually transposed [d][s] so the PV B-operand is a contiguous 32B load.
// -------------------------------------------------------------------------
__global__ __launch_bounds__(32) void flash_attn(
    const _Float16* __restrict__ Q, const _Float16* __restrict__ Ks,
    const _Float16* __restrict__ Vs, float* __restrict__ O)
{
  const int lane = threadIdx.x & 31;
  const int g    = lane >> 4;
  const int l15  = lane & 15;
  const int qt = blockIdx.x;
  const int h  = blockIdx.y;
  const int b  = blockIdx.z;
  const int kh = h >> 2;           // NH/NKV = 4
  const int q0 = qt * 16;

  __shared__ _Float16 k_lds[32][64];   // [key][d]  (TDM destination)
  __shared__ _Float16 vt_lds[64][32];  // [d][key]  (transposed V)
  __shared__ _Float16 p_lds[16][32];   // [qrow][key]

  // Load Q tile (16 x 64) into the two K=32 A-operands (16B runs + concat).
  v16h qa[2];
  {
    const _Float16* qrow = Q + ((size_t)(b * S_ + q0 + l15) * NH_ + h) * HD_;
    qa[0] = CAT16(*(const hv8*)(qrow + 16 * g),      *(const hv8*)(qrow + 8 + 16 * g));
    qa[1] = CAT16(*(const hv8*)(qrow + 32 + 16 * g), *(const hv8*)(qrow + 40 + 16 * g));
  }

#if USE_TDM
  // Static part of the K-tile D# (group1): workgroup_mask=0, data_size=2B,
  // tensor_dim0=64, tensor_dim1=S, tile_dim0=64, tile_dim1=32,
  // tensor_dim0_stride = NKV*HD elements.
  const i32x8 g1 = { (int)(1u << 16),
                     (int)(64u << 16),
                     (int)((unsigned)S_ << 16),
                     (int)(64u << 16),
                     32,
                     NKV_ * HD_,
                     0, 0 };
  const i32x4 gz4 = { 0, 0, 0, 0 };
  const i32x8 gz8 = { 0, 0, 0, 0, 0, 0, 0, 0 };
  const unsigned int lds_base = (unsigned int)(size_t)&k_lds[0][0];
#endif

  v8f acc[4] = {};
  float mrow[8], lrow[8];
#pragma unroll
  for (int v = 0; v < 8; ++v) { mrow[v] = -1e30f; lrow[v] = 0.f; }

  const int nchunks = (q0 + 16 + 31) >> 5;
  for (int c = 0; c < nchunks; ++c) {
    const int s0 = c * 32;

    // --- stage K via TDM (or manual fallback) ---
#if USE_TDM
    {
      const unsigned long long gaddr =
          (unsigned long long)(size_t)(Ks + ((size_t)(b * S_ + s0) * NKV_ + kh) * HD_);
      const u32x4 g0 = { 1u,                                  // count=1, user desc
                         lds_base,                            // lds_addr
                         (unsigned int)gaddr,                 // global_addr[31:0]
                         ((unsigned int)(gaddr >> 32) & 0x01FFFFFFu) |
                             (2u << 30) };                    // addr[56:32] | type=2
      __builtin_amdgcn_tensor_load_to_lds(g0, g1, gz4, gz4, gz8, 0);
    }
#else
    {
      const _Float16* krow = Ks + ((size_t)(b * S_ + s0 + lane) * NKV_ + kh) * HD_;
#pragma unroll
      for (int d = 0; d < 64; d += 8)
        *(hv8*)&k_lds[lane][d] = *(const hv8*)(krow + d);
    }
#endif

    // --- stage V transposed (pure f16 copy, scalar transpose stores) ---
    {
      const _Float16* vrow = Vs + ((size_t)(b * S_ + s0 + lane) * NKV_ + kh) * HD_;
#pragma unroll
      for (int d = 0; d < 64; d += 8) {
        const hv8 vv = *(const hv8*)(vrow + d);
        vt_lds[d + 0][lane] = vv[0]; vt_lds[d + 1][lane] = vv[1];
        vt_lds[d + 2][lane] = vv[2]; vt_lds[d + 3][lane] = vv[3];
        vt_lds[d + 4][lane] = vv[4]; vt_lds[d + 5][lane] = vv[5];
        vt_lds[d + 6][lane] = vv[6]; vt_lds[d + 7][lane] = vv[7];
      }
    }
#if USE_TDM
    __builtin_amdgcn_s_wait_tensorcnt(0);
#endif
    __syncthreads();

    // --- scores: S = Q K^T (B-operand = contiguous 32B from k_lds) ---
    v8f sc[2] = {};
#pragma unroll
    for (int j = 0; j < 2; ++j) {
#pragma unroll
      for (int cc = 0; cc < 2; ++cc) {
        const v16h kb = *(const v16h*)&k_lds[j * 16 + l15][cc * 32 + 16 * g];
        sc[j] = WMMA_F16(qa[cc], kb, sc[j]);
      }
    }

    // --- scale + causal mask (diagonal chunk only) + online softmax ---
    const bool diag = (c == nchunks - 1);
#pragma unroll
    for (int v = 0; v < 8; ++v) {
      float s0v = sc[0][v] * 0.125f;            // 1/sqrt(64)
      float s1v = sc[1][v] * 0.125f;
      if (diag) {
        const int qi = q0 + v + 8 * g;
        s0v = (s0 + l15      > qi) ? -1e30f : s0v;
        s1v = (s0 + 16 + l15 > qi) ? -1e30f : s1v;
      }
      float t  = group_max(fmaxf(s0v, s1v));
      float mn = fmaxf(mrow[v], t);
      const float alpha = __expf(mrow[v] - mn);
      mrow[v] = mn;
      const float p0 = __expf(s0v - mn);
      const float p1 = __expf(s1v - mn);
      lrow[v] = lrow[v] * alpha + group_sum(p0 + p1);
#pragma unroll
      for (int n = 0; n < 4; ++n) acc[n][v] *= alpha;
      p_lds[v + 8 * g][l15]      = (_Float16)p0;
      p_lds[v + 8 * g][16 + l15] = (_Float16)p1;
    }
    __syncthreads();

    // --- P -> A-layout (two 16B runs + concat), then out += P @ V ---
    const hv8 plo = *(const hv8*)&p_lds[l15][16 * g];
    const hv8 phi = *(const hv8*)&p_lds[l15][16 * g + 8];
    const v16h pa = CAT16(plo, phi);

#pragma unroll
    for (int n = 0; n < 4; ++n) {
      const v16h vb = *(const v16h*)&vt_lds[n * 16 + l15][16 * g];
      acc[n] = WMMA_F16(pa, vb, acc[n]);
    }
    __syncthreads();
  }

  // --- epilogue: divide by l, write 16x64 output tile (f32 for O-proj) ---
#pragma unroll
  for (int v = 0; v < 8; ++v) {
    const int r = v + 8 * g;
    const float inv = 1.0f / lrow[v];
    float* orow = O + ((size_t)(b * S_ + q0 + r) * NH_ + h) * HD_;
#pragma unroll
    for (int n = 0; n < 4; ++n)
      orow[n * 16 + l15] = acc[n][v] * inv;
  }
}

// -------------------------------------------------------------------------
// Launch: QKV projections (f16 out) -> RoPE(q,k) -> flash -> O projection.
// Workspace: q[4M halfs] k[1M halfs] v[1M halfs] ao[4M floats] = 28 MB.
// -------------------------------------------------------------------------
extern "C" void kernel_launch(void* const* d_in, const int* in_sizes, int n_in,
                              void* d_out, int out_size, void* d_ws, size_t ws_size,
                              hipStream_t stream) {
  const float* x    = (const float*)d_in[0];
  const float* cosT = (const float*)d_in[1];
  const float* sinT = (const float*)d_in[2];
  const float* wq   = (const float*)d_in[3];
  const float* wk   = (const float*)d_in[4];
  const float* wv   = (const float*)d_in[5];
  const float* wo   = (const float*)d_in[6];

  const int M = B_ * S_;                 // 4096
  _Float16* q  = (_Float16*)d_ws;
  _Float16* k  = q + (size_t)M * HID_;
  _Float16* v  = k + (size_t)M * (NKV_ * HD_);
  float*    ao = (float*)(v + (size_t)M * (NKV_ * HD_));

  // QKV projections (f16 outputs)
  gemm_wmma_f16<<<dim3(HID_ / 64, M / 64), 128, 0, stream>>>(x, wq, q, M, HID_, HID_, 1);
  gemm_wmma_f16<<<dim3((NKV_ * HD_) / 64, M / 64), 128, 0, stream>>>(x, wk, k, M, NKV_ * HD_, HID_, 1);
  gemm_wmma_f16<<<dim3((NKV_ * HD_) / 64, M / 64), 128, 0, stream>>>(x, wv, v, M, NKV_ * HD_, HID_, 1);

  // RoPE (in place, f16)
  {
    const int nq = M * NH_ * 32;
    const int nk = M * NKV_ * 32;
    rope_kernel<<<(nq + 255) / 256, 256, 0, stream>>>(q, cosT, sinT, NH_, nq);
    rope_kernel<<<(nk + 255) / 256, 256, 0, stream>>>(k, cosT, sinT, NKV_, nk);
  }

  // Flash attention: grid (q-tiles, heads, batch), one wave per block
  flash_attn<<<dim3(S_ / 16, NH_, B_), 32, 0, stream>>>(q, k, v, ao);

  // Output projection (f32 output)
  gemm_wmma_f16<<<dim3(HID_ / 64, M / 64), 128, 0, stream>>>(ao, wo, d_out, M, HID_, HID_, 0);
}
